// PopNetAudio_38465727103170
// MI455X (gfx1250) — compile-verified, matrix operand order checked by reading, not run on hardware
//
#include <hip/hip_runtime.h>

// ---------------------------------------------------------------------------
// PopNetAudio SNN forward, MI455X (gfx1250, wave32).
// FC layer on v_wmma_f32_16x16x32_bf16 with a 2x-bf16 weight split
// (w = w_hi + w_lo); spikes are exact {0,1} in bf16, fp32 accumulate.
// K padded to 496*32 so every wave runs a uniform 62-step K chunk
// (no exec-mask looping), and each wave computes two M-tiles sharing the
// B operands (4 WMMAs per 8 b128 loads).
// ---------------------------------------------------------------------------

typedef __bf16 bf16_t;
typedef __attribute__((ext_vector_type(16))) bf16_t v16bf;
typedef __attribute__((ext_vector_type(8)))  bf16_t v8bf;
typedef __attribute__((ext_vector_type(8)))  float  v8f;

union V16 { v16bf v; v8bf h[2]; };

#define THR   1.5f
#define BETA  0.9f

// sizes
#define BATCH   256
#define L1      1981            // (8000-80)/4+1
#define LP      495             // floor(1981/4)
#define L2      493             // 495-3+1
#define KDIM    15776           // 32*493 (valid K)
#define KPAD    15872           // 496*32  (padded K, zeros in pad)
#define KSTEPS  62              // 496/8 K-steps (of 32) per wave
#define HID     250
#define NSTEP   50

static constexpr size_t N_CUR1  = (size_t)BATCH * 16 * L1;   // 8,114,176
static constexpr size_t N_P     = (size_t)BATCH * 16 * LP;   // 2,027,520
static constexpr size_t N_MEM2  = (size_t)BATCH * 32 * L2;   // 4,038,656
static constexpr size_t N_MEM3  = (size_t)BATCH * 256;       // padded cols
static constexpr size_t N_SPK2P = (size_t)BATCH * KPAD;      // ushorts (padded)

__device__ __forceinline__ unsigned short f2bf(float f) {
    unsigned int u = __float_as_uint(f);
    u += 0x7FFFu + ((u >> 16) & 1u);        // round-to-nearest-even
    return (unsigned short)(u >> 16);
}
__device__ __forceinline__ float bf2f(unsigned short h) {
    return __uint_as_float(((unsigned int)h) << 16);
}

// ---------------------------------------------------------------------------
// Fold BN into conv weights/bias (runs once, tiny).
// ---------------------------------------------------------------------------
__global__ void prep_params_kernel(
    const float* __restrict__ c1w, const float* __restrict__ c1b,
    const float* __restrict__ g1,  const float* __restrict__ be1,
    const float* __restrict__ m1,  const float* __restrict__ v1,
    const float* __restrict__ c2w, const float* __restrict__ c2b,
    const float* __restrict__ g2,  const float* __restrict__ be2,
    const float* __restrict__ m2,  const float* __restrict__ v2,
    float* __restrict__ w1f, float* __restrict__ b1f,
    float* __restrict__ w2f, float* __restrict__ b2f)
{
    const int tid = threadIdx.x;
    for (int i = tid; i < 16 * 80; i += 256) {
        int oc = i / 80;
        w1f[i] = c1w[i] * (g1[oc] * rsqrtf(v1[oc] + 1e-5f));
    }
    for (int i = tid; i < 16; i += 256) {
        float inv = g1[i] * rsqrtf(v1[i] + 1e-5f);
        b1f[i] = c1b[i] * inv + be1[i] - m1[i] * inv;
    }
    for (int i = tid; i < 32 * 16 * 3; i += 256) {
        int oc = i / 48;
        w2f[i] = c2w[i] * (g2[oc] * rsqrtf(v2[oc] + 1e-5f));
    }
    for (int i = tid; i < 32; i += 256) {
        float inv = g2[i] * rsqrtf(v2[i] + 1e-5f);
        b2f[i] = c2b[i] * inv + be2[i] - m2[i] * inv;
    }
}

// ---------------------------------------------------------------------------
// conv1 (k=80, stride=4) + folded BN1 -> cur1 [256,16,1981]. Time invariant.
// ---------------------------------------------------------------------------
__global__ __launch_bounds__(256)
void conv1_bn_kernel(const float* __restrict__ x,
                     const float* __restrict__ w1f,
                     const float* __restrict__ b1f,
                     float* __restrict__ cur1)
{
    __shared__ float sw[16 * 80];
    __shared__ float sb[16];
    for (int i = threadIdx.x; i < 1280; i += 256) sw[i] = w1f[i];
    if (threadIdx.x < 16) sb[threadIdx.x] = b1f[threadIdx.x];
    __syncthreads();

    const size_t idx = (size_t)blockIdx.x * 256 + threadIdx.x;
    if (idx >= N_CUR1) return;
    const int pos = (int)(idx % L1);
    const int oc  = (int)((idx / L1) % 16);
    const int b   = (int)(idx / ((size_t)L1 * 16));

    const float* xb = x + (size_t)b * 8000 + (size_t)pos * 4;
    const float* wb = sw + oc * 80;
    float sum = sb[oc];
    #pragma unroll 8
    for (int k = 0; k < 80; ++k) sum += wb[k] * xb[k];
    cur1[idx] = sum;
}

// ---------------------------------------------------------------------------
// Split fc1_w (fp32 [250,15776]) into hi/lo bf16 with KPAD pitch; rows padded
// to 256 and K padded to 15872, pad entries = 0.
// ---------------------------------------------------------------------------
__global__ __launch_bounds__(256)
void split_w_kernel(const float* __restrict__ fc1_w,
                    unsigned short* __restrict__ whi,
                    unsigned short* __restrict__ wlo)
{
    const size_t idx = (size_t)blockIdx.x * 256 + threadIdx.x;
    if (idx >= N_SPK2P) return;
    const size_t n = idx / KPAD;
    const size_t k = idx % KPAD;
    unsigned short h = 0, l = 0;
    if (n < HID && k < KDIM) {
        float f = fc1_w[n * KDIM + k];
        h = f2bf(f);
        l = f2bf(f - bf2f(h));
    }
    whi[idx] = h;
    wlo[idx] = l;
}

// ---------------------------------------------------------------------------
// LIF1 (subtract-reset) on mem1[256,16,1981] + maxpool4 of binary spikes.
// pout==494 also updates the orphan position 1980.
// ---------------------------------------------------------------------------
__global__ __launch_bounds__(256)
void lif1_pool_kernel(float* __restrict__ mem1,
                      const float* __restrict__ cur1,
                      float* __restrict__ p)
{
    const size_t idx = (size_t)blockIdx.x * 256 + threadIdx.x;
    if (idx >= N_P) return;
    const int    pout = (int)(idx % LP);
    const size_t bc   = idx / LP;                         // b*16 + c
    const size_t base = bc * L1 + (size_t)pout * 4;

    float pmax = 0.0f;
    const int nup = (pout == LP - 1) ? 5 : 4;
    for (int i = 0; i < nup; ++i) {
        float mold  = mem1[base + i];
        float reset = (mold > THR) ? THR : 0.0f;
        float mnew  = BETA * mold + cur1[base + i] - reset;
        mem1[base + i] = mnew;
        if (i < 4) pmax = fmaxf(pmax, (mnew > THR) ? 1.0f : 0.0f);
    }
    p[idx] = pmax;
}

// ---------------------------------------------------------------------------
// conv2 (16->32, k=3) + folded BN2 + LIF2 -> spk2 as bf16 bits {0, 0x3F80}
// in [B][KPAD] layout (valid region = oc*493+pos < 15776).
// ---------------------------------------------------------------------------
__global__ __launch_bounds__(256)
void conv2_lif2_kernel(const float* __restrict__ p,
                       const float* __restrict__ w2f,
                       const float* __restrict__ b2f,
                       float* __restrict__ mem2,
                       unsigned short* __restrict__ spk2)
{
    __shared__ float sw[32 * 48];
    __shared__ float sb[32];
    for (int i = threadIdx.x; i < 1536; i += 256) sw[i] = w2f[i];
    if (threadIdx.x < 32) sb[threadIdx.x] = b2f[threadIdx.x];
    __syncthreads();

    const size_t idx = (size_t)blockIdx.x * 256 + threadIdx.x;
    if (idx >= N_MEM2) return;
    const int pos = (int)(idx % L2);
    const int oc  = (int)((idx / L2) % 32);
    const int b   = (int)(idx / ((size_t)L2 * 32));

    const float* pb = p + (size_t)b * 16 * LP + pos;
    const float* wb = sw + oc * 48;
    float sum = sb[oc];
    #pragma unroll
    for (int ic = 0; ic < 16; ++ic) {
        sum += wb[ic * 3 + 0] * pb[(size_t)ic * LP + 0]
             + wb[ic * 3 + 1] * pb[(size_t)ic * LP + 1]
             + wb[ic * 3 + 2] * pb[(size_t)ic * LP + 2];
    }
    float mold  = mem2[idx];
    float reset = (mold > THR) ? THR : 0.0f;
    float mnew  = BETA * mold + sum - reset;
    mem2[idx] = mnew;
    spk2[(size_t)b * KPAD + (size_t)oc * L2 + pos] =
        (mnew > THR) ? (unsigned short)0x3F80 : (unsigned short)0;
}

// ---------------------------------------------------------------------------
// FC via WMMA bf16 + LIF3.
// Block = (2 M-tiles, 1 N-tile); 8 waves split K uniformly (62 steps of 32
// each, KPAD=496*32). B operands (w_hi/w_lo) shared across the two M-tiles:
// 4 WMMAs per 8 b128 loads. LDS-reduce 8 K-partials, then LIF3 + writes.
// Operand layout per CDNA5 ISA 7.12.2: lanes 0-15 K{0..7,16..23},
// lanes 16-31 K{8..15,24..31}, 2 bf16 per VGPR.
// ---------------------------------------------------------------------------
__global__ __launch_bounds__(256)
void gemm_lif3_kernel(const unsigned short* __restrict__ spk2,
                      const unsigned short* __restrict__ whi,
                      const unsigned short* __restrict__ wlo,
                      const float* __restrict__ fc1_b,
                      float* __restrict__ mem3,
                      float* __restrict__ out_hid,   // d_out + 128000
                      int t)
{
    __shared__ float red[2][8][256];   // 16 KB

    const int lane = threadIdx.x & 31;
    const int wave = threadIdx.x >> 5;
    const int mT0  = blockIdx.x * 2;       // batch tiles {mT0, mT0+1}
    const int nT   = blockIdx.y;           // neuron tile 0..15 (256 padded)
    const int half = lane >> 4;            // K-half selector
    const int l16  = lane & 15;

    const size_t kb = (size_t)wave * KSTEPS * 32;   // this wave's K origin

    const unsigned short* aP0 = spk2 + ((size_t)(mT0 * 16 + l16)) * KPAD + kb + half * 8;
    const unsigned short* aP1 = aP0 + (size_t)16 * KPAD;
    const unsigned short* bhP = whi  + ((size_t)(nT * 16 + l16)) * KPAD + kb + half * 8;
    const unsigned short* blP = wlo  + ((size_t)(nT * 16 + l16)) * KPAD + kb + half * 8;

    v8f acc0 = {};
    v8f acc1 = {};
    #pragma unroll 2
    for (int kk = 0; kk < KSTEPS; ++kk) {
        const int k0 = kk * 32;
        V16 a0, a1, bh, bl;
        a0.h[0] = *reinterpret_cast<const v8bf*>(aP0 + k0);
        a0.h[1] = *reinterpret_cast<const v8bf*>(aP0 + k0 + 16);
        a1.h[0] = *reinterpret_cast<const v8bf*>(aP1 + k0);
        a1.h[1] = *reinterpret_cast<const v8bf*>(aP1 + k0 + 16);
        bh.h[0] = *reinterpret_cast<const v8bf*>(bhP + k0);
        bh.h[1] = *reinterpret_cast<const v8bf*>(bhP + k0 + 16);
        bl.h[0] = *reinterpret_cast<const v8bf*>(blP + k0);
        bl.h[1] = *reinterpret_cast<const v8bf*>(blP + k0 + 16);
        acc0 = __builtin_amdgcn_wmma_f32_16x16x32_bf16(
                   false, a0.v, false, bh.v, (short)0, acc0, false, false);
        acc0 = __builtin_amdgcn_wmma_f32_16x16x32_bf16(
                   false, a0.v, false, bl.v, (short)0, acc0, false, false);
        acc1 = __builtin_amdgcn_wmma_f32_16x16x32_bf16(
                   false, a1.v, false, bh.v, (short)0, acc1, false, false);
        acc1 = __builtin_amdgcn_wmma_f32_16x16x32_bf16(
                   false, a1.v, false, bl.v, (short)0, acc1, false, false);
    }

    // stash partials: element id = lane*8 + v
    #pragma unroll
    for (int v = 0; v < 8; ++v) {
        red[0][wave][lane * 8 + v] = acc0[v];
        red[1][wave][lane * 8 + v] = acc1[v];
    }
    __syncthreads();

    // reduce 8 K-partials; tid owns one element of each of the two tiles
    const int tid    = threadIdx.x;
    const int lane_r = tid >> 3;
    const int v_r    = tid & 7;
    const int n      = nT * 16 + (lane_r & 15);
    const int mloc   = (lane_r < 16) ? v_r : v_r + 8;

    if (n < HID) {
        const float bias = fc1_b[n];
        #pragma unroll
        for (int which = 0; which < 2; ++which) {
            float sum = red[which][0][tid];
            #pragma unroll
            for (int w = 1; w < 8; ++w) sum += red[which][w][tid];

            const int m = (mT0 + which) * 16 + mloc;
            const size_t midx = (size_t)m * 256 + n;
            float mold  = mem3[midx];
            float reset = (mold > THR) ? THR : 0.0f;
            float mnew  = BETA * mold + (sum + bias) - reset;
            mem3[midx] = mnew;
            out_hid[((size_t)t * BATCH + m) * HID + n] = (mnew > THR) ? 1.0f : 0.0f;
        }
    }
}

// ---------------------------------------------------------------------------
// Population readout: out[t,b,cls] = sum over the 25 neurons of each class.
// ---------------------------------------------------------------------------
__global__ __launch_bounds__(256)
void popsum_kernel(const float* __restrict__ hid, float* __restrict__ out)
{
    const size_t idx = (size_t)blockIdx.x * 256 + threadIdx.x;
    if (idx >= (size_t)NSTEP * BATCH * 10) return;
    const int    cls = (int)(idx % 10);
    const size_t tb  = idx / 10;                       // t*256 + b
    const float* h = hid + tb * HID + cls * 25;
    float s = 0.0f;
    #pragma unroll
    for (int j = 0; j < 25; ++j) s += h[j];
    out[idx] = s;
}

// ---------------------------------------------------------------------------
extern "C" void kernel_launch(void* const* d_in, const int* in_sizes, int n_in,
                              void* d_out, int out_size, void* d_ws, size_t ws_size,
                              hipStream_t stream)
{
    (void)in_sizes; (void)n_in; (void)out_size; (void)ws_size;

    const float* x       = (const float*)d_in[0];
    const float* conv1_w = (const float*)d_in[1];
    const float* conv1_b = (const float*)d_in[2];
    const float* bn1_g   = (const float*)d_in[3];
    const float* bn1_be  = (const float*)d_in[4];
    const float* bn1_m   = (const float*)d_in[5];
    const float* bn1_v   = (const float*)d_in[6];
    const float* conv2_w = (const float*)d_in[7];
    const float* conv2_b = (const float*)d_in[8];
    const float* bn2_g   = (const float*)d_in[9];
    const float* bn2_be  = (const float*)d_in[10];
    const float* bn2_m   = (const float*)d_in[11];
    const float* bn2_v   = (const float*)d_in[12];
    const float* fc1_w   = (const float*)d_in[13];
    const float* fc1_b   = (const float*)d_in[14];

    float* out = (float*)d_out;                 // [50,256,10] ++ [50,256,250]

    // workspace carve-out (256B aligned)
    char* ws = (char*)d_ws;
    size_t o = 0;
    auto carve = [&](size_t bytes) -> char* {
        char* r = ws + o;
        o += (bytes + 255) & ~(size_t)255;
        return r;
    };
    float* cur1 = (float*)carve(N_CUR1 * 4);
    float* mem1 = (float*)carve(N_CUR1 * 4);
    float* p    = (float*)carve(N_P * 4);
    float* mem2 = (float*)carve(N_MEM2 * 4);
    float* mem3 = (float*)carve(N_MEM3 * 4);
    float* w1f  = (float*)carve(16 * 80 * 4);
    float* b1f  = (float*)carve(16 * 4);
    float* w2f  = (float*)carve(32 * 48 * 4);
    float* b2f  = (float*)carve(32 * 4);
    unsigned short* spk2 = (unsigned short*)carve(N_SPK2P * 2);
    unsigned short* whi  = (unsigned short*)carve(N_SPK2P * 2);
    unsigned short* wlo  = (unsigned short*)carve(N_SPK2P * 2);

    // zero membrane state + spk2 (so K-pad region stays zero); capture-legal
    hipMemsetAsync(mem1, 0, N_CUR1 * 4, stream);
    hipMemsetAsync(mem2, 0, N_MEM2 * 4, stream);
    hipMemsetAsync(mem3, 0, N_MEM3 * 4, stream);
    hipMemsetAsync(spk2, 0, N_SPK2P * 2, stream);

    prep_params_kernel<<<1, 256, 0, stream>>>(
        conv1_w, conv1_b, bn1_g, bn1_be, bn1_m, bn1_v,
        conv2_w, conv2_b, bn2_g, bn2_be, bn2_m, bn2_v,
        w1f, b1f, w2f, b2f);

    conv1_bn_kernel<<<(unsigned)((N_CUR1 + 255) / 256), 256, 0, stream>>>(
        x, w1f, b1f, cur1);

    split_w_kernel<<<(unsigned)((N_SPK2P + 255) / 256), 256, 0, stream>>>(
        fc1_w, whi, wlo);

    float* out_hid = out + (size_t)NSTEP * BATCH * 10;   // + 128000

    for (int t = 0; t < NSTEP; ++t) {
        lif1_pool_kernel<<<(unsigned)((N_P + 255) / 256), 256, 0, stream>>>(
            mem1, cur1, p);
        conv2_lif2_kernel<<<(unsigned)((N_MEM2 + 255) / 256), 256, 0, stream>>>(
            p, w2f, b2f, mem2, spk2);
        gemm_lif3_kernel<<<dim3(8, 16), 256, 0, stream>>>(
            spk2, whi, wlo, fc1_b, mem3, out_hid, t);
    }

    popsum_kernel<<<(unsigned)(((size_t)NSTEP * BATCH * 10 + 255) / 256), 256, 0, stream>>>(
        out_hid, out);
}
